// LlamaAttention_86251533238384
// MI455X (gfx1250) — matrix operand order repaired and checked by reading, not compiled, not run on hardware
//
#include <hip/hip_runtime.h>
#include <hip/hip_bf16.h>

// ---------------------------------------------------------------------------
// LlamaAttention prefill for MI455X (gfx1250).
// bf16 WMMA + fp32 accumulate; async-to-LDS double-buffered B staging for the
// projection GEMMs; causal flash attention with online softmax.
// B=2, S=2048, H=4096, NH=32, NKV=8, HD=128, N_REP=4.
// ---------------------------------------------------------------------------

typedef unsigned short u16;
typedef __attribute__((ext_vector_type(16))) __bf16 v16bf;
typedef __attribute__((ext_vector_type(8)))  float  v8f;

union FragU { v16bf v; uint4 q[2]; };

static constexpr int BB   = 2;
static constexpr int SS   = 2048;
static constexpr int HH   = 4096;
static constexpr int NH   = 32;
static constexpr int NKV  = 8;
static constexpr int HD   = 128;
static constexpr int LDB_LDS = 40;   // 32-elem rows padded to 40 (80B) for banks
static constexpr float SM_SCALE = 0.08838834764831845f;     // 1/sqrt(128)
static constexpr float NEG_LN_ROPE = -0.14391156831212787f; // -ln(10000)/64

__device__ __forceinline__ u16 f2bf(float f) {
  unsigned int u = __float_as_uint(f);
  return (u16)((u + 0x7FFFu + ((u >> 16) & 1u)) >> 16);  // RNE
}

__device__ __forceinline__ v8f wmma_bf16(v16bf a, v16bf b, v8f c) {
  return __builtin_amdgcn_wmma_f32_16x16x32_bf16(false, a, false, b, (short)0, c,
                                                 false, false);
}

// A fragment: 16(M) x 32(K) bf16 tile from row-major matrix, leading dim lda.
__device__ __forceinline__ v16bf load_a_frag(const u16* A, int lda, int mBase,
                                             int kBase, int lane) {
  FragU f;
  int m  = mBase + (lane & 15);
  int hl = lane >> 4;
  const u16* p = A + (size_t)m * lda + kBase + hl * 8;
  f.q[0] = *(const uint4*)(p);        // K = kBase + hl*8 + 0..7
  f.q[1] = *(const uint4*)(p + 16);   // K = kBase + 16 + hl*8 + 0..7
  return f.v;
}

// B fragment: 32(K) x 16(N), source N-major: element (k,n) at B[n*ldb + k].
__device__ __forceinline__ v16bf load_b_frag(const u16* B, int ldb, int nBase,
                                             int kBase, int lane) {
  FragU f;
  int n  = nBase + (lane & 15);
  int hl = lane >> 4;
  const u16* p = B + (size_t)n * ldb + kBase + hl * 8;
  f.q[0] = *(const uint4*)(p);
  f.q[1] = *(const uint4*)(p + 16);
  return f.v;
}

// B fragment from padded LDS tile (ldb = LDB_LDS, kBase = 0).
__device__ __forceinline__ v16bf load_b_frag_lds(const u16* buf, int nBase,
                                                 int lane) {
  FragU f;
  int n  = nBase + (lane & 15);
  int hl = lane >> 4;
  const u16* p = buf + n * LDB_LDS + hl * 8;
  f.q[0] = *(const uint4*)(p);
  f.q[1] = *(const uint4*)(p + 16);
  return f.v;
}

// ---- CDNA5 async global->LDS copy (ASYNCcnt path, ISA §10 / §15.18) -------
__device__ __forceinline__ unsigned lds_byte_addr(void* p) {
  return (unsigned)(unsigned long long)(__attribute__((address_space(3))) void*)p;
}
__device__ __forceinline__ void async_ld_b128(u16* ldsp, const u16* gaddr) {
  unsigned loff = lds_byte_addr(ldsp);
  asm volatile("global_load_async_to_lds_b128 %0, %1, off"
               :: "v"(loff), "v"(gaddr) : "memory");
}
__device__ __forceinline__ void wait_async_le4() {
  asm volatile("s_wait_asynccnt 0x4" ::: "memory");
}
__device__ __forceinline__ void wait_async_le0() {
  asm volatile("s_wait_asynccnt 0x0" ::: "memory");
}

// Stage one 128(N) x 32(K) bf16 B tile into padded LDS: 512 x 16B pieces,
// 4 async b128 per thread (4 per-wave ASYNCcnt increments).
__device__ __forceinline__ void stage_b_tile(const u16* Wstrip, int k0,
                                             u16* buf, int tid) {
  #pragma unroll
  for (int j = 0; j < 4; ++j) {
    int piece = tid + 128 * j;            // 0..511
    int row   = piece >> 2;               // 0..127
    int seg   = (piece & 3) * 8;          // element offset 0,8,16,24
    async_ld_b128(buf + row * LDB_LDS + seg,
                  Wstrip + (size_t)row * HH + k0 + seg);
  }
}

// Strip GEMM: c[8] (16 rows x 128 cols) = A[row0..row0+16) x Wstrip^T.
// 4 waves per block share the B strip via double-buffered async LDS staging.
__device__ __forceinline__ void gemm_strip(const u16* A, int row0,
                                           const u16* Wstrip, u16* buf0,
                                           u16* buf1, int lane, int tid,
                                           v8f c[8]) {
  stage_b_tile(Wstrip, 0, buf0, tid);
  for (int i = 0; i < HH / 32; ++i) {
    u16* cur = (i & 1) ? buf1 : buf0;
    __syncthreads();                       // everyone done reading buf[i^1]
    if (i + 1 < HH / 32) {
      stage_b_tile(Wstrip, (i + 1) * 32, (i & 1) ? buf0 : buf1, tid);
      wait_async_le4();                    // chunk i's 4 copies landed (ours)
    } else {
      wait_async_le0();
    }
    __syncthreads();                       // chunk i landed for all waves
    v16bf a = load_a_frag(A, HH, row0, i * 32, lane);
    #pragma unroll
    for (int cb = 0; cb < 8; ++cb)
      c[cb] = wmma_bf16(a, load_b_frag_lds(cur, cb * 16, lane), c[cb]);
  }
}

// ---------------------------------------------------------------------------
// Kernel 0a: fp32 -> bf16 cast (hidden states), 4 elems/thread.
// ---------------------------------------------------------------------------
__global__ void cast_f32_to_bf16(const float* __restrict__ in,
                                 u16* __restrict__ out, int n4) {
  int i = blockIdx.x * blockDim.x + threadIdx.x;
  if (i >= n4) return;
  float4 f = ((const float4*)in)[i];
  ushort4 o;
  o.x = f2bf(f.x); o.y = f2bf(f.y); o.z = f2bf(f.z); o.w = f2bf(f.w);
  ((ushort4*)out)[i] = o;
}

// ---------------------------------------------------------------------------
// Kernel 0b: tiled transpose+cast: R x C row-major fp32 -> C x R bf16.
// ---------------------------------------------------------------------------
__global__ void transpose_f32_to_bf16(const float* __restrict__ in,
                                      u16* __restrict__ out, int R, int C) {
  __shared__ float tile[32][33];
  int c0 = blockIdx.x * 32, r0 = blockIdx.y * 32;
  #pragma unroll
  for (int i = threadIdx.y; i < 32; i += 8)
    tile[i][threadIdx.x] = in[(size_t)(r0 + i) * C + c0 + threadIdx.x];
  __syncthreads();
  #pragma unroll
  for (int i = threadIdx.y; i < 32; i += 8)
    out[(size_t)(c0 + i) * R + r0 + threadIdx.x] = f2bf(tile[threadIdx.x][i]);
}

// ---------------------------------------------------------------------------
// Kernel 1: fused QKV projection + RoPE (async-LDS staged B).
// Block = 4 waves, consecutive 16-token row tiles, same 128-wide head strip.
// strips: 0..31 = Q heads, 32..39 = K heads, 40..47 = V heads.
// ---------------------------------------------------------------------------
__global__ __launch_bounds__(128) void qkv_rope_kernel(
    const u16* __restrict__ hs, const u16* __restrict__ wq_t,
    const u16* __restrict__ wk_t, const u16* __restrict__ wv_t,
    const int* __restrict__ posids, u16* __restrict__ Qo,
    u16* __restrict__ Ko, u16* __restrict__ Vt) {
  __shared__ u16 bstage[2][128 * LDB_LDS];   // 2 x 10 KB

  int tid  = threadIdx.x;
  int lane = tid & 31;
  int gw = blockIdx.x * 4 + (tid >> 5);
  int rt = gw & 255;       // 256 row tiles over B*S
  int strip = gw >> 8;     // 48 strips (same for all 4 waves of the block)

  const u16* Wt; int head, kind;
  if (strip < 32)      { kind = 0; head = strip;      Wt = wq_t; }
  else if (strip < 40) { kind = 1; head = strip - 32; Wt = wk_t; }
  else                 { kind = 2; head = strip - 40; Wt = wv_t; }

  int token0 = rt * 16;
  v8f c[8] = {};
  gemm_strip(hs, token0, Wt + (size_t)head * HD * HH, bstage[0], bstage[1],
             lane, tid, c);

  int b  = token0 >> 11;
  int s0 = token0 & (SS - 1);
  int hl = lane >> 4, n = lane & 15;

  if (kind < 2) {
    // RoPE: partner of dim d is d +/- 64 -> frag cb and cb+4, same lane slot.
    #pragma unroll
    for (int cb = 0; cb < 4; ++cb) {
      float invf = __expf((float)(cb * 16 + n) * NEG_LN_ROPE);
      #pragma unroll
      for (int r = 0; r < 8; ++r) {
        int tok = s0 + 8 * hl + r;
        float th = (float)posids[b * SS + tok] * invf;
        float sn, cs;
        __sincosf(th, &sn, &cs);
        float x0 = c[cb][r], x1 = c[cb + 4][r];
        c[cb][r]     = x0 * cs - x1 * sn;
        c[cb + 4][r] = x1 * cs + x0 * sn;
      }
    }
    u16* dst = (kind == 0) ? Qo : Ko;
    int nh   = (kind == 0) ? NH : NKV;
    size_t base = ((size_t)(b * nh + head) * SS + s0) * HD;
    #pragma unroll
    for (int cb = 0; cb < 8; ++cb)
      #pragma unroll
      for (int r = 0; r < 8; ++r)
        dst[base + (size_t)(8 * hl + r) * HD + cb * 16 + n] = f2bf(c[cb][r]);
  } else {
    size_t base = (size_t)(b * NKV + head) * HD * SS;   // V transposed [hd, s]
    #pragma unroll
    for (int cb = 0; cb < 8; ++cb)
      #pragma unroll
      for (int r = 0; r < 8; ++r)
        Vt[base + (size_t)(cb * 16 + n) * SS + s0 + 8 * hl + r] = f2bf(c[cb][r]);
  }
}

// ---------------------------------------------------------------------------
// Kernel 2: causal flash attention, bf16 WMMA, fp32 online softmax.
// One wave per (b, head, 16-query tile); 32 keys per iteration.
// ---------------------------------------------------------------------------
__global__ __launch_bounds__(128) void attn_kernel(
    const u16* __restrict__ Q, const u16* __restrict__ K,
    const u16* __restrict__ Vt, u16* __restrict__ ctx) {
  __shared__ u16 pbuf[4][16 * 32];      // per-wave P-tile staging (1 KB each)

  int lane = threadIdx.x & 31;
  int wv   = threadIdx.x >> 5;
  int gw   = blockIdx.x * 4 + wv;
  int qt = gw & 127;                    // S/16 query tiles
  int h  = (gw >> 7) & 31;              // head
  int b  = gw >> 12;                    // batch
  int kvh = h >> 2;                     // GQA: 4 Q heads per KV head
  int hl = lane >> 4, n = lane & 15;

  const u16* Qb = Q  + (((size_t)(b * NH + h) * SS) + qt * 16) * HD;
  const u16* Kb = K  + (size_t)(b * NKV + kvh) * SS * HD;
  const u16* Vb = Vt + (size_t)(b * NKV + kvh) * HD * SS;

  v16bf qf[4];
  #pragma unroll
  for (int c = 0; c < 4; ++c) qf[c] = load_a_frag(Qb, HD, 0, 32 * c, lane);

  v8f o[8] = {};
  float mrow[8], lrow[8];
  #pragma unroll
  for (int r = 0; r < 8; ++r) { mrow[r] = -1e30f; lrow[r] = 0.f; }

  int qend = qt * 16 + 16;
  for (int k0 = 0; k0 < qend; k0 += 32) {
    // ---- S = Q @ K^T for 32 keys (two 16-key column tiles) ----
    v8f s1 = {}, s2 = {};
    #pragma unroll
    for (int c = 0; c < 4; ++c) {
      s1 = wmma_bf16(qf[c], load_b_frag(Kb, HD, k0,      32 * c, lane), s1);
      s2 = wmma_bf16(qf[c], load_b_frag(Kb, HD, k0 + 16, 32 * c, lane), s2);
    }
    // ---- scale + causal mask + online softmax ----
    #pragma unroll
    for (int r = 0; r < 8; ++r) {
      int qrow = qt * 16 + 8 * hl + r;
      float v1 = s1[r] * SM_SCALE; if (k0 + n      > qrow) v1 = -1e30f;
      float v2 = s2[r] * SM_SCALE; if (k0 + 16 + n > qrow) v2 = -1e30f;

      float mx = fmaxf(v1, v2);
      #pragma unroll
      for (int off = 1; off < 16; off <<= 1)
        mx = fmaxf(mx, __shfl_xor(mx, off, 32));

      float mnew = fmaxf(mrow[r], mx);
      float corr = __expf(mrow[r] - mnew);
      float p1 = __expf(v1 - mnew);
      float p2 = __expf(v2 - mnew);
      float rs = p1 + p2;
      #pragma unroll
      for (int off = 1; off < 16; off <<= 1)
        rs += __shfl_xor(rs, off, 32);

      lrow[r] = lrow[r] * corr + rs;
      mrow[r] = mnew;
      s1[r] = p1; s2[r] = p2;
      #pragma unroll
      for (int cb = 0; cb < 8; ++cb) o[cb][r] *= corr;
    }
    // ---- re-layout P (C-frag) -> A-frag via per-wave LDS tile ----
    u16* pb = pbuf[wv];
    #pragma unroll
    for (int r = 0; r < 8; ++r) {
      pb[(8 * hl + r) * 32 + n]      = f2bf(s1[r]);
      pb[(8 * hl + r) * 32 + 16 + n] = f2bf(s2[r]);
    }
    asm volatile("s_wait_dscnt 0" ::: "memory");
    FragU pf;
    {
      const u16* p = &pb[(lane & 15) * 32 + hl * 8];
      pf.q[0] = *(const uint4*)(p);
      pf.q[1] = *(const uint4*)(p + 16);
    }
    // ---- O += P @ V (Vt is [hd, s] so B pairs are contiguous) ----
    #pragma unroll
    for (int cb = 0; cb < 8; ++cb)
      o[cb] = wmma_bf16(pf.v, load_b_frag(Vb, SS, cb * 16, k0, lane), o[cb]);
  }

  // ---- normalize + write context [b, s, h*HD + d] bf16 ----
  float invl[8];
  #pragma unroll
  for (int r = 0; r < 8; ++r) invl[r] = 1.0f / lrow[r];
  size_t base = ((size_t)b * SS + qt * 16) * (NH * HD) + h * HD;
  #pragma unroll
  for (int cb = 0; cb < 8; ++cb)
    #pragma unroll
    for (int r = 0; r < 8; ++r)
      ctx[base + (size_t)(8 * hl + r) * (NH * HD) + cb * 16 + n] =
          f2bf(o[cb][r] * invl[r]);
}

// ---------------------------------------------------------------------------
// Kernel 3: output projection (async-LDS staged B, same strip structure).
// One wave -> 16 rows x 128 cols; 4 waves share the Wo strip.
// ---------------------------------------------------------------------------
__global__ __launch_bounds__(128) void oproj_kernel(
    const u16* __restrict__ ctx, const u16* __restrict__ wo_t,
    float* __restrict__ out) {
  __shared__ u16 bstage[2][128 * LDB_LDS];

  int tid  = threadIdx.x;
  int lane = tid & 31;
  int gw = blockIdx.x * 4 + (tid >> 5);
  int rt = gw & 255;            // 256 row tiles
  int strip = gw >> 8;          // 32 strips of 128 output cols
  int hl = lane >> 4, n = lane & 15;

  v8f c[8] = {};
  gemm_strip(ctx, rt * 16, wo_t + (size_t)strip * 128 * HH, bstage[0],
             bstage[1], lane, tid, c);

  #pragma unroll
  for (int cb = 0; cb < 8; ++cb)
    #pragma unroll
    for (int r = 0; r < 8; ++r)
      out[(size_t)(rt * 16 + 8 * hl + r) * HH + strip * 128 + cb * 16 + n] =
          c[cb][r];
}

// ---------------------------------------------------------------------------
extern "C" void kernel_launch(void* const* d_in, const int* in_sizes, int n_in,
                              void* d_out, int out_size, void* d_ws,
                              size_t ws_size, hipStream_t stream) {
  const float* hs  = (const float*)d_in[0];   // [2, 2048, 4096]
  const int*   pos = (const int*)d_in[1];     // [2, 2048]
  // d_in[2]: attention_mask — causal, applied analytically in attn_kernel.
  const float* Wq  = (const float*)d_in[3];   // [4096, 4096]
  const float* Wk  = (const float*)d_in[4];   // [4096, 1024]
  const float* Wv  = (const float*)d_in[5];   // [4096, 1024]
  const float* Wo  = (const float*)d_in[6];   // [4096, 4096]
  float* out = (float*)d_out;                 // [2, 2048, 4096]

  // workspace layout (u16 elements, all 16B aligned)
  u16* w = (u16*)d_ws;
  const size_t NBIG = (size_t)BB * SS * HH;       // 16,777,216
  const size_t NKVW = (size_t)HH * NKV * HD;      //  4,194,304
  u16* hs_bf = w;  w += NBIG;
  u16* wq_t  = w;  w += NBIG;
  u16* wk_t  = w;  w += NKVW;
  u16* wv_t  = w;  w += NKVW;
  u16* wo_t  = w;  w += NBIG;
  u16* Qb    = w;  w += NBIG;                     // [b, nh, s, hd]
  u16* Kb    = w;  w += NKVW;                     // [b, nkv, s, hd]
  u16* Vt    = w;  w += NKVW;                     // [b, nkv, hd, s]
  u16* ctx   = w;                                 // [b, s, nh*hd]

  // 0) casts + weight transposes
  cast_f32_to_bf16<<<(int)(NBIG / 4 / 256), 256, 0, stream>>>(hs, hs_bf,
                                                              (int)(NBIG / 4));
  dim3 tb(32, 8);
  transpose_f32_to_bf16<<<dim3(128, 128), tb, 0, stream>>>(Wq, wq_t, HH, 4096);
  transpose_f32_to_bf16<<<dim3(32, 128),  tb, 0, stream>>>(Wk, wk_t, HH, 1024);
  transpose_f32_to_bf16<<<dim3(32, 128),  tb, 0, stream>>>(Wv, wv_t, HH, 1024);
  transpose_f32_to_bf16<<<dim3(128, 128), tb, 0, stream>>>(Wo, wo_t, HH, 4096);

  // 1) QKV + RoPE: 256 row tiles * 48 strips = 12288 waves, 4 waves/block
  qkv_rope_kernel<<<3072, 128, 0, stream>>>(hs_bf, wq_t, wk_t, wv_t, pos,
                                            Qb, Kb, Vt);

  // 2) flash attention: 2 * 32 * 128 = 8192 waves, 4 waves/block
  attn_kernel<<<2048, 128, 0, stream>>>(Qb, Kb, Vt, ctx);

  // 3) output projection: 256 row tiles * 32 strips = 8192 waves, 4/block
  oproj_kernel<<<2048, 128, 0, stream>>>(ctx, wo_t, out);
}
